// My_model_v3_graclus_2layer_dropout_SAGE_4277787427383
// MI455X (gfx1250) — compile-verified, matrix operand order checked by reading, not run on hardware
//
#include <hip/hip_runtime.h>
#include <hip/hip_bf16.h>

// ---------------------------------------------------------------------------
// Types for CDNA5 WMMA (wave32): A/B = 16 x bf16 (8 VGPRs), C/D = 8 x f32
// ---------------------------------------------------------------------------
typedef __attribute__((ext_vector_type(16))) __bf16 v16bf;
typedef __attribute__((ext_vector_type(8)))  float  v8f;

// Model constants
#define BATCH   128
#define NDRUG   40
#define DFEAT   77
#define DD      128
#define N0      706
#define CFEAT   3
#define DC      256
#define FC_CELL 1024
#define OUT_CELL 512
#define REGD    640   // OUT_CELL + 128

enum { ACT_NONE = 0, ACT_RELU = 1, ACT_ELU = 2 };

__device__ __forceinline__ unsigned f32_to_bf16_bits(float f) {
    unsigned u = __float_as_uint(f);
    return (u + 0x7FFFu + ((u >> 16) & 1u)) >> 16;   // RNE
}

// Decompose k (0..31) into the ISA 16-bit 16x32 fragment slot.
__device__ __forceinline__ void frag_slot(int k, int& vv, int& hi) {
    int grp = k >> 4;          // K 0-15 vs 16-31
    int rem = k & 15;
    hi = rem >> 3;             // lanes 0-15 vs 16-31
    int vin = (rem >> 1) & 3;
    vv = grp * 4 + vin;
}

// ---------------------------------------------------------------------------
// Generic GEMM:  C = act( A(MxK,f32) * B(KxN,f32 weights) [+ bias] [+ Cin] )
// bf16 WMMA, f32 accumulate. Block = 256 threads = 8 waves; tile 64x128.
// Wave (wm,wn) in 2x4 grid -> 32x32 output = 4 WMMA accumulators.
// LDS tiles stored in fragment order: lane L owns 16 contiguous bf16 per
// subtile -> fragment load = one 32B per-lane load (2x ds_load_b128).
// All model GEMMs have M%64==0 and N%128==0; only K can be ragged (77, 3),
// so staging/epilogue take a single uniform fast/slow branch.
// ---------------------------------------------------------------------------
#define BM 64
#define BN 128
#define BK 32
#define SUB_U 256   // subtile = 512 shorts = 256 uints (32 lanes x 8 uints)

__global__ __launch_bounds__(256) void k_gemm(
    const float* __restrict__ A, int lda,
    const float* __restrict__ Bw, int ldb,
    const float* __restrict__ bias,
    float* __restrict__ C, int ldc,
    const float* __restrict__ Cin,
    int M, int N, int K, int act)
{
    __shared__ unsigned int As32[4 * SUB_U];   // 4 A-subtiles (16x32)  = 4 KB
    __shared__ unsigned int Bs32[8 * SUB_U];   // 8 B-subtiles (32x16)  = 8 KB

    const int tid  = threadIdx.x;
    const int lane = tid & 31;
    const int wave = tid >> 5;
    const int hiL  = lane >> 4;
    const int lm   = lane & 15;
    const int wm   = wave >> 2;     // 2 wave-rows  (32 rows each)
    const int wn   = wave & 3;      // 4 wave-cols  (32 cols each)
    const int m0   = blockIdx.y * BM;
    const int n0   = blockIdx.x * BN;
    const bool fullMN = (m0 + BM <= M) && (n0 + BN <= N);

    v8f c00 = {}; v8f c01 = {}; v8f c10 = {}; v8f c11 = {};

    for (int kc = 0; kc < K; kc += BK) {
        const bool full = fullMN && (kc + BK <= K);
        if (full) {
            // ---- fast path: unguarded b64 loads ----
            #pragma unroll
            for (int it = 0; it < 4; ++it) {            // A: 1024 k-pairs
                int p = it * 256 + tid;
                int r = p >> 4, k = (p & 15) << 1;
                float2 v = *(const float2*)&A[(size_t)(m0 + r) * lda + (kc + k)];
                int vv, hi; frag_slot(k, vv, hi);
                As32[(r >> 4) * SUB_U + (hi * 16 + (r & 15)) * 8 + vv] =
                    f32_to_bf16_bits(v.x) | (f32_to_bf16_bits(v.y) << 16);
            }
            unsigned short* BsS = (unsigned short*)Bs32;
            #pragma unroll
            for (int it = 0; it < 8; ++it) {            // B: 2048 n-pairs
                int p = it * 256 + tid;
                int k = p >> 6, n = (p & 63) << 1;
                float2 v = *(const float2*)&Bw[(size_t)(kc + k) * ldb + (n0 + n)];
                int vv, hi; frag_slot(k, vv, hi);
                int lo = k & 1;
                int base = (n >> 4) * 512 + (hi * 16 + (n & 15)) * 16 + 2 * vv + lo;
                BsS[base]      = (unsigned short)f32_to_bf16_bits(v.x);
                BsS[base + 16] = (unsigned short)f32_to_bf16_bits(v.y);
            }
        } else {
            // ---- slow path: guarded (K remainder / ragged tiles) ----
            #pragma unroll
            for (int it = 0; it < 4; ++it) {
                int p = it * 256 + tid;
                int r = p >> 4, k = (p & 15) << 1;
                int gr = m0 + r, gk = kc + k;
                float v0 = (gr < M && gk     < K) ? A[(size_t)gr * lda + gk]     : 0.0f;
                float v1 = (gr < M && gk + 1 < K) ? A[(size_t)gr * lda + gk + 1] : 0.0f;
                int vv, hi; frag_slot(k, vv, hi);
                As32[(r >> 4) * SUB_U + (hi * 16 + (r & 15)) * 8 + vv] =
                    f32_to_bf16_bits(v0) | (f32_to_bf16_bits(v1) << 16);
            }
            #pragma unroll
            for (int it = 0; it < 8; ++it) {
                int p = it * 256 + tid;
                int kp = p >> 7, n = p & 127;
                int k = kp << 1;
                int gk = kc + k, gn = n0 + n;
                float v0 = (gk     < K && gn < N) ? Bw[(size_t)gk * ldb + gn]       : 0.0f;
                float v1 = (gk + 1 < K && gn < N) ? Bw[(size_t)(gk + 1) * ldb + gn] : 0.0f;
                int vv, hi; frag_slot(k, vv, hi);
                Bs32[(n >> 4) * SUB_U + (hi * 16 + (n & 15)) * 8 + vv] =
                    f32_to_bf16_bits(v0) | (f32_to_bf16_bits(v1) << 16);
            }
        }
        __syncthreads();

        // prefetch next K-chunk (gfx1250 global_prefetch_b8)
        if (kc + BK < K) {
            int pr = m0 + (tid >> 3); if (pr >= M) pr = M - 1;
            __builtin_prefetch(&A[(size_t)pr * lda + (kc + BK)], 0, 1);
            int bk = kc + BK + (tid >> 7); if (bk >= K) bk = K - 1;
            int bn = n0 + (tid & 127); if (bn >= N) bn = N - 1;
            __builtin_prefetch(&Bw[(size_t)bk * ldb + bn], 0, 1);
        }

        // ---- fragments: one 32B contiguous per-lane load each ----
        v16bf a0 = *(const v16bf*)&As32[(wm * 2 + 0) * SUB_U + lane * 8];
        v16bf a1 = *(const v16bf*)&As32[(wm * 2 + 1) * SUB_U + lane * 8];
        v16bf b0 = *(const v16bf*)&Bs32[(wn * 2 + 0) * SUB_U + lane * 8];
        v16bf b1 = *(const v16bf*)&Bs32[(wn * 2 + 1) * SUB_U + lane * 8];

        c00 = __builtin_amdgcn_wmma_f32_16x16x32_bf16(false, a0, false, b0, (short)0, c00, false, false);
        c01 = __builtin_amdgcn_wmma_f32_16x16x32_bf16(false, a0, false, b1, (short)0, c01, false, false);
        c10 = __builtin_amdgcn_wmma_f32_16x16x32_bf16(false, a1, false, b0, (short)0, c10, false, false);
        c11 = __builtin_amdgcn_wmma_f32_16x16x32_bf16(false, a1, false, b1, (short)0, c11, false, false);
        __syncthreads();
    }

    // ---- epilogue: element i of accumulator -> row = base + i + 8*hiL ----
    const int rowBase = m0 + wm * 32;
    const int colBase = n0 + wn * 32;
    v8f accs[4] = { c00, c01, c10, c11 };
    if (fullMN) {
        #pragma unroll
        for (int q = 0; q < 4; ++q) {
            int rs = rowBase + (q >> 1) * 16;
            int cs = colBase + (q & 1) * 16 + lm;
            #pragma unroll
            for (int i = 0; i < 8; ++i) {
                int r = rs + i + 8 * hiL;
                float x = accs[q][i];
                if (Cin)  x += Cin[(size_t)r * ldc + cs];
                if (bias) x += bias[cs];
                if (act == ACT_RELU)     x = fmaxf(x, 0.0f);
                else if (act == ACT_ELU) x = (x > 0.0f) ? x : expm1f(x);
                C[(size_t)r * ldc + cs] = x;
            }
        }
    } else {
        #pragma unroll
        for (int q = 0; q < 4; ++q) {
            int rs = rowBase + (q >> 1) * 16;
            int cs = colBase + (q & 1) * 16 + lm;
            #pragma unroll
            for (int i = 0; i < 8; ++i) {
                int r = rs + i + 8 * hiL;
                if (r < M && cs < N) {
                    float x = accs[q][i];
                    if (Cin)  x += Cin[(size_t)r * ldc + cs];
                    if (bias) x += bias[cs];
                    if (act == ACT_RELU)     x = fmaxf(x, 0.0f);
                    else if (act == ACT_ELU) x = (x > 0.0f) ? x : expm1f(x);
                    C[(size_t)r * ldc + cs] = x;
                }
            }
        }
    }
}

// ---------------------------------------------------------------------------
// Support kernels
// ---------------------------------------------------------------------------
__global__ void k_fill(float* p, float v, size_t n) {
    size_t i = (size_t)blockIdx.x * blockDim.x + threadIdx.x;
    size_t stride = (size_t)gridDim.x * blockDim.x;
    for (; i < n; i += stride) p[i] = v;
}

__global__ void k_copy_pad(const float* __restrict__ src, int srcw,
                           float* __restrict__ dst, int dstw, int rows) {
    size_t i = (size_t)blockIdx.x * blockDim.x + threadIdx.x;
    size_t n = (size_t)rows * dstw;
    if (i >= n) return;
    int r = (int)(i / dstw), c = (int)(i % dstw);
    dst[i] = (c < srcw) ? src[(size_t)r * srcw + c] : 0.0f;
}

__global__ void k_copy(const float* __restrict__ src, float* __restrict__ dst, size_t n) {
    size_t i = (size_t)blockIdx.x * blockDim.x + threadIdx.x;
    if (i < n) dst[i] = src[i];
}

__global__ void k_scatter_add(const int* __restrict__ ei, int E,
                              const float* __restrict__ x, int ldx,
                              float* __restrict__ agg, int ldagg,
                              float* __restrict__ deg, int F) {
    size_t i = (size_t)blockIdx.x * blockDim.x + threadIdx.x;
    size_t n = (size_t)E * F;
    if (i >= n) return;
    int e = (int)(i / F), f = (int)(i % F);
    int src = ei[e], dst = ei[E + e];
    atomicAdd(&agg[(size_t)dst * ldagg + f], x[(size_t)src * ldx + f]);
    if (f == 0 && deg) atomicAdd(&deg[dst], 1.0f);
}

__global__ void k_div_deg(float* __restrict__ agg, int ldagg,
                          const float* __restrict__ deg, int rows, int F) {
    size_t i = (size_t)blockIdx.x * blockDim.x + threadIdx.x;
    size_t n = (size_t)rows * F;
    if (i >= n) return;
    int r = (int)(i / F), f = (int)(i % F);
    agg[(size_t)r * ldagg + f] /= fmaxf(deg[r], 1.0f);
}

// segment-max over clusters; values are post-ReLU (>=0) so int atomicMax works
__global__ void k_pool_max(const float* __restrict__ x, int ldx,
                           const int* __restrict__ clus,
                           float* __restrict__ out, int ldo, int rows, int F) {
    size_t i = (size_t)blockIdx.x * blockDim.x + threadIdx.x;
    size_t n = (size_t)rows * F;
    if (i >= n) return;
    int r = (int)(i / F), f = (int)(i % F);
    float v = x[(size_t)r * ldx + f];
    atomicMax((int*)&out[(size_t)clus[r] * ldo + f], __float_as_int(v));
}

__global__ __launch_bounds__(256) void k_bn_stats(const float* __restrict__ x, int ldx,
                                                  int rows, int F,
                                                  float* __restrict__ sum,
                                                  float* __restrict__ sq) {
    int col = threadIdx.x;
    if (col >= F) return;
    int r0 = blockIdx.x * 64;
    int r1 = r0 + 64; if (r1 > rows) r1 = rows;
    float s = 0.0f, q = 0.0f;
    for (int r = r0; r < r1; ++r) {
        float v = x[(size_t)r * ldx + col];
        s += v; q += v * v;
    }
    atomicAdd(&sum[col], s);
    atomicAdd(&sq[col], q);
}

__global__ void k_bn_apply(const float* __restrict__ x, int ldx,
                           float* __restrict__ y, int ldy, int rows, int F,
                           const float* __restrict__ sum, const float* __restrict__ sq,
                           const float* __restrict__ g, const float* __restrict__ b) {
    size_t i = (size_t)blockIdx.x * blockDim.x + threadIdx.x;
    size_t n = (size_t)rows * F;
    if (i >= n) return;
    int r = (int)(i / F), f = (int)(i % F);
    float inv = 1.0f / (float)rows;
    float m = sum[f] * inv;
    float var = sq[f] * inv - m * m;
    float v = x[(size_t)r * ldx + f];
    y[(size_t)r * ldy + f] = g[f] * (v - m) * rsqrtf(var + 1e-5f) + b[f];
}

__global__ void k_drug_pool(const float* __restrict__ x, float* __restrict__ xdg) {
    int i = blockIdx.x * blockDim.x + threadIdx.x;
    if (i >= BATCH * DD) return;
    int b = i / DD, f = i % DD;
    float s = 0.0f, mx = -3.4e38f;
    for (int n = 0; n < NDRUG; ++n) {
        float v = x[(size_t)(b * NDRUG + n) * DD + f];
        s += v; mx = fmaxf(mx, v);
    }
    xdg[(size_t)b * (2 * DD) + f]      = s * (1.0f / NDRUG);
    xdg[(size_t)b * (2 * DD) + DD + f] = mx;
}

__global__ void k_concat(const float* __restrict__ xd, const float* __restrict__ xc,
                         float* __restrict__ z) {
    int i = blockIdx.x * blockDim.x + threadIdx.x;
    if (i >= BATCH * REGD) return;
    int b = i / REGD, c = i % REGD;
    z[i] = (c < 128) ? xd[(size_t)b * 128 + c] : xc[(size_t)b * OUT_CELL + (c - 128)];
}

__global__ __launch_bounds__(256) void k_final(const float* __restrict__ z,
                                               const float* __restrict__ w3,
                                               const float* __restrict__ b3,
                                               float* __restrict__ out) {
    __shared__ float red[256];
    int b = blockIdx.x, t = threadIdx.x;
    float s = 0.0f;
    for (int c = t; c < REGD; c += 256) s += z[(size_t)b * REGD + c] * w3[c];
    red[t] = s;
    __syncthreads();
    for (int off = 128; off > 0; off >>= 1) {
        if (t < off) red[t] += red[t + off];
        __syncthreads();
    }
    if (t == 0) out[b] = red[0] + b3[0];
}

// ---------------------------------------------------------------------------
// Host orchestration
// ---------------------------------------------------------------------------
static inline void gemm(hipStream_t s, const float* A, int lda, const float* B, int ldb,
                        const float* bias, float* C, int ldc, const float* Cin,
                        int M, int N, int K, int act) {
    dim3 g((N + BN - 1) / BN, (M + BM - 1) / BM);
    k_gemm<<<g, 256, 0, s>>>(A, lda, B, ldb, bias, C, ldc, Cin, M, N, K, act);
}
static inline void fill(hipStream_t s, float* p, float v, size_t n) {
    int blocks = (int)((n + 1023) / 1024); if (blocks > 65535) blocks = 65535;
    k_fill<<<blocks, 256, 0, s>>>(p, v, n);
}
static inline dim3 g1(size_t n) { return dim3((unsigned)((n + 255) / 256)); }

extern "C" void kernel_launch(void* const* d_in, const int* in_sizes, int n_in,
                              void* d_out, int out_size, void* d_ws, size_t ws_size,
                              hipStream_t stream) {
    const float* drug_x  = (const float*)d_in[0];
    const int*   drug_ei = (const int*)d_in[1];
    const float* cell_x  = (const float*)d_in[3];
    const int*   cell_ei[3] = { (const int*)d_in[4], (const int*)d_in[5], (const int*)d_in[6] };
    const int*   clus[3]    = { (const int*)d_in[7], (const int*)d_in[8], (const int*)d_in[9] };
    const int Ed = in_sizes[1] / 2;
    const int Ec[3] = { in_sizes[4] / 2, in_sizes[5] / 2, in_sizes[6] / 2 };

    auto P = [&](int i) { return (const float*)d_in[10 + i]; };
    // gin{i}: w1 b1 w2 b2 g b -> 0..17 ; demb: w b -> 18..19
    // sage{i}: wl bl wr g b  -> 20..34 ; cemb: w1 b1 w2 b2 -> 35..38
    // reg: w1 b1 w2 b2 w3 b3 -> 39..44

    float* w = (float*)d_ws;
    size_t o = 0;
    auto alloc = [&](size_t n) { float* p = w + o; o += n; return p; };
    const int NDN = BATCH * NDRUG;
    float* dx  = alloc((size_t)NDN * DD);
    float* dh  = alloc((size_t)NDN * DD);
    float* dt1 = alloc((size_t)NDN * DD);
    float* dt2 = alloc((size_t)NDN * DD);
    float* bn_sum = alloc(DC);
    float* bn_sq  = alloc(DC);
    float* xdg    = alloc((size_t)BATCH * 2 * DD);
    float* x_drug = alloc((size_t)BATCH * DD);
    const int NC0 = BATCH * 706, NC1 = BATCH * 353, NC2 = BATCH * 177, NC3 = BATCH * 89;
    float* cagg = alloc((size_t)NC0 * DC);
    float* ct   = alloc((size_t)NC0 * DC);
    float* cxc  = alloc((size_t)NC1 * DC);
    float* deg  = alloc((size_t)NC0);
    float* hcell  = alloc((size_t)BATCH * FC_CELL);
    float* x_cell = alloc((size_t)BATCH * OUT_CELL);
    float* z  = alloc((size_t)BATCH * REGD);
    float* z1 = alloc((size_t)BATCH * REGD);
    float* z2 = alloc((size_t)BATCH * REGD);
    (void)ws_size; (void)n_in;

    // ================= DRUG BRANCH (GIN x3) =================
    k_copy_pad<<<g1((size_t)NDN * DD), 256, 0, stream>>>(drug_x, DFEAT, dx, DD, NDN);
    for (int i = 0; i < 3; ++i) {
        const float* w1 = P(i * 6 + 0); const float* b1 = P(i * 6 + 1);
        const float* w2 = P(i * 6 + 2); const float* b2 = P(i * 6 + 3);
        const float* gg = P(i * 6 + 4); const float* gb = P(i * 6 + 5);
        const int din = (i == 0) ? DFEAT : DD;
        k_copy<<<g1((size_t)NDN * DD), 256, 0, stream>>>(dx, dh, (size_t)NDN * DD);
        k_scatter_add<<<g1((size_t)Ed * din), 256, 0, stream>>>(drug_ei, Ed, dx, DD, dh, DD, nullptr, din);
        gemm(stream, dh, DD, w1, DD, b1, dt1, DD, nullptr, NDN, DD, din, ACT_RELU);
        gemm(stream, dt1, DD, w2, DD, b2, dt2, DD, nullptr, NDN, DD, DD, ACT_RELU);
        fill(stream, bn_sum, 0.0f, DD); fill(stream, bn_sq, 0.0f, DD);
        k_bn_stats<<<(NDN + 63) / 64, 256, 0, stream>>>(dt2, DD, NDN, DD, bn_sum, bn_sq);
        k_bn_apply<<<g1((size_t)NDN * DD), 256, 0, stream>>>(dt2, DD, dx, DD, NDN, DD, bn_sum, bn_sq, gg, gb);
    }
    k_drug_pool<<<g1((size_t)BATCH * DD), 256, 0, stream>>>(dx, xdg);
    gemm(stream, xdg, 2 * DD, P(18), DD, P(19), x_drug, DD, nullptr, BATCH, DD, 2 * DD, ACT_RELU);

    // ================= CELL BRANCH (SAGE x3 + graclus pool) =================
    const float* xc_ptr = cell_x; int xc_ld = CFEAT;
    const int nns[3]   = { NC0, NC1, NC2 };
    const int nouts[3] = { NC1, NC2, NC3 };
    const int dins[3]  = { CFEAT, DC, DC };
    for (int i = 0; i < 3; ++i) {
        const float* wl = P(20 + i * 5 + 0); const float* bl = P(20 + i * 5 + 1);
        const float* wr = P(20 + i * 5 + 2);
        const float* gg = P(20 + i * 5 + 3); const float* gb = P(20 + i * 5 + 4);
        const int nn = nns[i], din = dins[i], nout = nouts[i];
        fill(stream, cagg, 0.0f, (size_t)nn * din);
        fill(stream, deg, 0.0f, (size_t)nn);
        k_scatter_add<<<g1((size_t)Ec[i] * din), 256, 0, stream>>>(cell_ei[i], Ec[i], xc_ptr, xc_ld, cagg, din, deg, din);
        k_div_deg<<<g1((size_t)nn * din), 256, 0, stream>>>(cagg, din, deg, nn, din);
        gemm(stream, cagg, din, wl, DC, bl, ct, DC, nullptr, nn, DC, din, ACT_NONE);
        gemm(stream, xc_ptr, xc_ld, wr, DC, nullptr, ct, DC, ct, nn, DC, din, ACT_RELU);
        fill(stream, cxc, 0.0f, (size_t)nout * DC);
        k_pool_max<<<g1((size_t)nn * DC), 256, 0, stream>>>(ct, DC, clus[i], cxc, DC, nn, DC);
        fill(stream, bn_sum, 0.0f, DC); fill(stream, bn_sq, 0.0f, DC);
        k_bn_stats<<<(nout + 63) / 64, 256, 0, stream>>>(cxc, DC, nout, DC, bn_sum, bn_sq);
        k_bn_apply<<<g1((size_t)nout * DC), 256, 0, stream>>>(cxc, DC, cxc, DC, nout, DC, bn_sum, bn_sq, gg, gb);
        xc_ptr = cxc; xc_ld = DC;
    }
    gemm(stream, cxc, 89 * DC, P(35), FC_CELL, P(36), hcell, FC_CELL, nullptr, BATCH, FC_CELL, 89 * DC, ACT_RELU);
    gemm(stream, hcell, FC_CELL, P(37), OUT_CELL, P(38), x_cell, OUT_CELL, nullptr, BATCH, OUT_CELL, FC_CELL, ACT_RELU);

    // ================= REGRESSION HEAD =================
    k_concat<<<g1((size_t)BATCH * REGD), 256, 0, stream>>>(x_drug, x_cell, z);
    gemm(stream, z,  REGD, P(39), REGD, P(40), z1, REGD, nullptr, BATCH, REGD, REGD, ACT_ELU);
    gemm(stream, z1, REGD, P(41), REGD, P(42), z2, REGD, nullptr, BATCH, REGD, REGD, ACT_ELU);
    k_final<<<BATCH, 256, 0, stream>>>(z2, P(43), P(44), (float*)d_out);
    (void)out_size;
}
